// DenseCaps_23519240912904
// MI455X (gfx1250) — matrix operand order repaired and checked by reading, not compiled, not run on hardware
//
#include <hip/hip_runtime.h>
#include <hip/hip_bf16.h>

typedef __attribute__((ext_vector_type(16))) _Float16 v16h;
typedef __attribute__((ext_vector_type(8)))  _Float16 v8h;
typedef __attribute__((ext_vector_type(4)))  _Float16 v4h;
typedef __attribute__((ext_vector_type(8)))  float    v8f;
typedef unsigned int u32x4 __attribute__((ext_vector_type(4)));
typedef int          i32x8 __attribute__((ext_vector_type(8)));
typedef int          i32x4 __attribute__((ext_vector_type(4)));

// Problem constants: B=32, I=4096, O=32, pose 4x4.
// GEMM view of votes-sum: M = 4*B = 128 (b,x), N = 4*O = 128 (o,z), K = 4*I = 16384 (i,y)
// P flat:   poses[b][i][x][y] -> b*65536 + i*16 + x*4 + y
// W flat:   w[o][i][y][z]     -> o*65536 + i*16 + y*4 + z
// S layout: [b][o][x][z]      -> b*512 + o*16 + x*4 + z   (16384 floats)
// ws: S at [0,16384), Vsum at [16384,32768)

__global__ void caps_zero(float* __restrict__ ws) {
    const int idx = blockIdx.x * blockDim.x + threadIdx.x;
    if (idx < 32768) ws[idx] = 0.0f;
}

// ---------------------------------------------------------------------------
// Kernel 1: s0 partials via WMMA.  grid.x = 64 K-chunks (256 k = 64 i each),
// block = 256 (8 waves).  Each wave owns a 16-row M-block, sweeps 8 N-tiles.
// A/B chunks converted f32->f16 and staged in LDS with fragment-friendly
// layouts: As[m][k] (k contiguous), Bs[n][k] (k contiguous).
// ---------------------------------------------------------------------------
__global__ __launch_bounds__(256)
void caps_votes_gemm(const float* __restrict__ P, const float* __restrict__ W,
                     float* __restrict__ S) {
    extern __shared__ _Float16 smem[];
    _Float16* As = smem;              // [128][256] halves = 64 KB
    _Float16* Bs = smem + 128 * 256;  // [128][256] halves = 64 KB

    const int tid  = threadIdx.x;
    const int lane = tid & 31;
    const int wave = tid >> 5;
    const int ibase = blockIdx.x * 64;          // 64 input capsules per chunk

    // ---- stage A (poses): As[m][4*ip + y] ----
    for (int t = tid; t < 128 * 64; t += 256) {
        const int m  = t >> 6;
        const int ip = t & 63;
        const int i  = ibase + ip;
        const float4 v = *(const float4*)(P + (size_t)(m >> 2) * 65536 +
                                          (size_t)i * 16 + (m & 3) * 4);
        __builtin_prefetch(W + (size_t)(m >> 2) * 65536 + (size_t)i * 16, 0, 1);
        v4h h = { (_Float16)v.x, (_Float16)v.y, (_Float16)v.z, (_Float16)v.w };
        *(v4h*)(As + m * 256 + ip * 4) = h;
    }
    // ---- stage B (weights): Bs[4*o+z][4*ip + y] ----
    for (int t = tid; t < 32 * 64 * 4; t += 256) {
        const int o  = t >> 8;
        const int r  = t & 255;
        const int ip = r >> 2;
        const int y  = r & 3;
        const int i  = ibase + ip;
        const float4 v = *(const float4*)(W + (size_t)o * 65536 +
                                          (size_t)i * 16 + y * 4);
        const int k = ip * 4 + y;
        Bs[(4 * o + 0) * 256 + k] = (_Float16)v.x;
        Bs[(4 * o + 1) * 256 + k] = (_Float16)v.y;
        Bs[(4 * o + 2) * 256 + k] = (_Float16)v.z;
        Bs[(4 * o + 3) * 256 + k] = (_Float16)v.w;
    }
    __syncthreads();

    const int Mbase = wave * 16;
    const int mrow  = Mbase + (lane & 15);
    // f16 A 16x32 lane layout: lanes 0-15 hold K[0..7]+K[16..23], lanes 16-31 K[8..15]+K[24..31]
    const int asel = (lane < 16) ? 0 : 8;
    // f16 B 32x16 lane layout: lanes 0-15 hold K[0..7]+K[16..23], lanes 16-31 K[8..15]+K[24..31] (col n = lane&15)
    const int bsel = (lane < 16) ? 0 : 8;

    v8f acc[8] = {};

#pragma unroll
    for (int ks = 0; ks < 8; ++ks) {            // 8 steps of K=32 -> 256 k per chunk
        const int k0 = ks * 32;
        const v8h alo = *(const v8h*)(As + mrow * 256 + k0 + asel);
        const v8h ahi = *(const v8h*)(As + mrow * 256 + k0 + 16 + asel);
        const v16h afrag = __builtin_shufflevector(alo, ahi,
            0, 1, 2, 3, 4, 5, 6, 7, 8, 9, 10, 11, 12, 13, 14, 15);
#pragma unroll
        for (int nt = 0; nt < 8; ++nt) {
            const int n = nt * 16 + (lane & 15);
            const v8h blo = *(const v8h*)(Bs + n * 256 + k0 + bsel);
            const v8h bhi = *(const v8h*)(Bs + n * 256 + k0 + 16 + bsel);
            const v16h bfrag = __builtin_shufflevector(blo, bhi,
                0, 1, 2, 3, 4, 5, 6, 7, 8, 9, 10, 11, 12, 13, 14, 15);
            acc[nt] = __builtin_amdgcn_wmma_f32_16x16x32_f16(
                false, afrag, false, bfrag, (short)0, acc[nt], false, false);
        }
    }

    // C/D 16x16 f32 layout: VGPR v -> M = v (lanes 0-15) or v+8 (lanes 16-31), N = lane&15
#pragma unroll
    for (int nt = 0; nt < 8; ++nt) {
#pragma unroll
        for (int v = 0; v < 8; ++v) {
            const int mc = Mbase + v + ((lane < 16) ? 0 : 8);
            const int nc = nt * 16 + (lane & 15);
            const int sidx = (mc >> 2) * 512 + (nc >> 2) * 16 + (mc & 3) * 4 + (nc & 3);
            atomicAdd(S + sidx, acc[nt][v]);
        }
    }
}

// ---------------------------------------------------------------------------
// Kernel 2: squash S[b,o] (scaled), then either accumulate into Vsum (mode 0)
// or emit final poses + activations (mode 1).  Re-zeroes S for the next pass.
// One thread per (b,o) capsule: 1024 threads.
// ---------------------------------------------------------------------------
__global__ __launch_bounds__(256)
void caps_squash(float* __restrict__ S, float* __restrict__ Vsum,
                 float* __restrict__ out, float scale, int mode) {
    const int idx = blockIdx.x * blockDim.x + threadIdx.x;
    if (idx >= 1024) return;
    float s[16];
    float n2 = 0.0f;
#pragma unroll
    for (int e = 0; e < 16; ++e) {
        s[e] = S[idx * 16 + e] * scale;
        n2 += s[e] * s[e];
        S[idx * 16 + e] = 0.0f;                 // reset accumulator for next pass
    }
    const float n = sqrtf(n2 + 1e-12f);
    const float f = n2 / ((1.0f + n2) * n);
    if (mode == 0) {
#pragma unroll
        for (int e = 0; e < 16; ++e) Vsum[idx * 16 + e] += s[e] * f;
    } else {
        float a2 = 0.0f;
#pragma unroll
        for (int e = 0; e < 16; ++e) {
            const float v = s[e] * f;
            out[idx * 16 + e] = v;
            a2 += v * v;
        }
        out[16384 + idx] = sqrtf(a2 + 1e-12f);
    }
}

// ---------------------------------------------------------------------------
// Kernel 3: one routing refinement pass.  grid = (16 i-chunks, 32 batches),
// block = 256 threads (one input capsule i per thread).
// Vsum[b] (2 KB) is staged global->LDS via the Tensor Data Mover (D# built per
// CDNA5 ISA sec.8; 1-D tile, data_size=4B), overlapping the
// DMA with the pose loads / sPart zeroing; completion via s_wait_tensorcnt.
// Then: recompute votes, logits = <vote, Vsum>, softmax over o, and a wave32
// butterfly + ds_add_f32 reduction of c*vote into S[b,o].
// ---------------------------------------------------------------------------
__global__ __launch_bounds__(256)
void caps_route(const float* __restrict__ P, const float* __restrict__ W,
                const float* __restrict__ Vsum, float* __restrict__ S) {
    __shared__ float sV[512];          // Vsum for this batch (filled by TDM)
    __shared__ float sPart[512];       // partial S for this batch
    __shared__ float sC[32 * 256];     // per-thread logit/coupling column

    const int tid  = threadIdx.x;
    const int lane = tid & 31;
    const int b    = blockIdx.y;
    const int i    = blockIdx.x * 256 + tid;

    sPart[tid]       = 0.0f;
    sPart[tid + 256] = 0.0f;

    if (tid < 32) {
        // ---- TDM descriptor: copy 512 f32 from Vsum + b*512 into sV ----
        const unsigned long long gaddr = (unsigned long long)(const void*)(Vsum + (size_t)b * 512);
        const unsigned int lds_off = (unsigned int)(unsigned long long)(const void*)sV; // LDS aperture: low 32 bits = LDS byte offset
        u32x4 g0;
        g0[0] = 1u;                                        // count=1, user mode, no gather
        g0[1] = lds_off;                                   // lds_addr (bytes)
        g0[2] = (unsigned int)(gaddr & 0xFFFFFFFFu);       // global_addr[31:0]
        g0[3] = (unsigned int)((gaddr >> 32) & 0x1FFFFFFu) // global_addr[56:32]
              | (2u << 30);                                // type = 2 ("image")
        i32x8 g1;
        g1[0] = (int)(2u << 16);                           // data_size = 4 bytes; wg_mask=0
        g1[1] = (int)((512u & 0xFFFFu) << 16);             // tensor_dim0[15:0] = 512
        g1[2] = 0;                                         // tensor_dim0[31:16]=0, tensor_dim1[15:0]=0
        g1[3] = (int)(512u << 16);                         // tile_dim0 = 512
        g1[4] = 0;                                         // tile_dim1 = 0 -> 1-D tile
        g1[5] = 512;                                       // tensor_dim0_stride = 512
        g1[6] = 0;
        g1[7] = 0;
        const i32x4 gz4 = {0, 0, 0, 0};                    // groups 2/3 unused (<=2-D)
        const i32x8 gz8 = {0, 0, 0, 0, 0, 0, 0, 0};        // extra group (6-arg variant)
        __builtin_amdgcn_tensor_load_to_lds(g0, g1, gz4, gz4, gz8, 0);
        __builtin_amdgcn_s_wait_tensorcnt(0);
    }

    float p[16];
#pragma unroll
    for (int e = 0; e < 16; e += 4) {
        const float4 v = *(const float4*)(P + (size_t)b * 65536 + (size_t)i * 16 + e);
        p[e] = v.x; p[e + 1] = v.y; p[e + 2] = v.z; p[e + 3] = v.w;
    }
    __syncthreads();   // publishes TDM-filled sV + zeroed sPart to all waves

    // --- logits per output capsule ---
    for (int o = 0; o < 32; ++o) {
        float wv[16];
#pragma unroll
        for (int e = 0; e < 16; e += 4) {
            const float4 v = *(const float4*)(W + (size_t)o * 65536 + (size_t)i * 16 + e);
            wv[e] = v.x; wv[e + 1] = v.y; wv[e + 2] = v.z; wv[e + 3] = v.w;
        }
        float l = 0.0f;
#pragma unroll
        for (int x = 0; x < 4; ++x)
#pragma unroll
            for (int z = 0; z < 4; ++z) {
                const float vv = p[x * 4 + 0] * wv[0 * 4 + z] + p[x * 4 + 1] * wv[1 * 4 + z]
                               + p[x * 4 + 2] * wv[2 * 4 + z] + p[x * 4 + 3] * wv[3 * 4 + z];
                l += vv * sV[o * 16 + x * 4 + z];
            }
        sC[o * 256 + tid] = l;
    }

    // --- softmax over o (private column, no barrier needed) ---
    float mx = -3.0e38f;
    for (int o = 0; o < 32; ++o) mx = fmaxf(mx, sC[o * 256 + tid]);
    float sum = 0.0f;
    for (int o = 0; o < 32; ++o) {
        const float e = __expf(sC[o * 256 + tid] - mx);
        sC[o * 256 + tid] = e;
        sum += e;
    }
    const float rs = 1.0f / sum;

    // --- accumulate c * vote into S partials ---
    for (int o = 0; o < 32; ++o) {
        const float c = sC[o * 256 + tid] * rs;
        float wv[16];
#pragma unroll
        for (int e = 0; e < 16; e += 4) {
            const float4 v = *(const float4*)(W + (size_t)o * 65536 + (size_t)i * 16 + e);
            wv[e] = v.x; wv[e + 1] = v.y; wv[e + 2] = v.z; wv[e + 3] = v.w;
        }
#pragma unroll
        for (int x = 0; x < 4; ++x)
#pragma unroll
            for (int z = 0; z < 4; ++z) {
                float vv = c * (p[x * 4 + 0] * wv[0 * 4 + z] + p[x * 4 + 1] * wv[1 * 4 + z]
                              + p[x * 4 + 2] * wv[2 * 4 + z] + p[x * 4 + 3] * wv[3 * 4 + z]);
                // wave32 butterfly reduction over the 32 lanes (32 input caps)
                vv += __shfl_xor(vv, 1);
                vv += __shfl_xor(vv, 2);
                vv += __shfl_xor(vv, 4);
                vv += __shfl_xor(vv, 8);
                vv += __shfl_xor(vv, 16);
                if (lane == x * 4 + z) atomicAdd(&sPart[o * 16 + x * 4 + z], vv);
            }
    }
    __syncthreads();
    atomicAdd(S + b * 512 + tid,       sPart[tid]);
    atomicAdd(S + b * 512 + 256 + tid, sPart[tid + 256]);
}

extern "C" void kernel_launch(void* const* d_in, const int* in_sizes, int n_in,
                              void* d_out, int out_size, void* d_ws, size_t ws_size,
                              hipStream_t stream) {
    const float* P = (const float*)d_in[0];   // input_poses [32,4096,4,4]
    // d_in[1] = input_activations: unused by the reference computation
    const float* W = (const float*)d_in[2];   // w [32,4096,4,4]
    float* S    = (float*)d_ws;               // [32,32,4,4] accumulator
    float* Vsum = S + 16384;                  // accumulated squashed poses
    float* out  = (float*)d_out;              // 16384 poses + 1024 activations

    caps_zero<<<128, 256, 0, stream>>>(S);

    // iter 0: uniform coupling -> pure GEMM, scale 1/32 folded into squash
    caps_votes_gemm<<<64, 256, 128 * 1024, stream>>>(P, W, S);
    caps_squash<<<4, 256, 0, stream>>>(S, Vsum, out, 1.0f / 32.0f, 0);

    // iter 1
    caps_route<<<dim3(16, 32), 256, 0, stream>>>(P, W, Vsum, S);
    caps_squash<<<4, 256, 0, stream>>>(S, Vsum, out, 1.0f, 0);

    // iter 2 (final)
    caps_route<<<dim3(16, 32), 256, 0, stream>>>(P, W, Vsum, S);
    caps_squash<<<4, 256, 0, stream>>>(S, Vsum, out, 1.0f, 1);
}
